// GCN_43722767073906
// MI455X (gfx1250) — matrix operand order repaired and checked by reading, not compiled, not run on hardware
//
#include <hip/hip_runtime.h>

// GCNConv(768->256, self-loops, sym-norm) + ReLU + global_mean_pool + Linear(256->5) + log_softmax
// MI455X / gfx1250, wave32. Main GEMM runs on v_wmma_f32_16x16x32_bf16.
// Workspace layout (floats): h[100k*256] | agg[100k*256] | deg[100k] | dinv[100k] | pooled[G*256] | cnt[G]
// (~207 MB total; everything we read is rewritten every call -> deterministic under graph replay).

#define IN_DIM 768
#define HID 256
#define N_CLASSES 5

typedef __attribute__((ext_vector_type(16))) __bf16          v16bf;
typedef __attribute__((ext_vector_type(8)))  float           v8f;
typedef __attribute__((ext_vector_type(16))) unsigned short  v16us;
typedef __attribute__((ext_vector_type(8)))  unsigned short  v8us;

static __device__ __forceinline__ unsigned short f2bf(float f) {
  unsigned u = __float_as_uint(f);
  u += 0x7FFFu + ((u >> 16) & 1u);   // round-to-nearest-even
  return (unsigned short)(u >> 16);
}

// ---------------- init: deg=1 (self-loop), cnt=0, pooled=0 ----------------
__global__ void init_k(float* __restrict__ deg, float* __restrict__ cnt,
                       float* __restrict__ pooled, int n, int g, int ph) {
  int i = blockIdx.x * blockDim.x + threadIdx.x;
  if (i < n)  deg[i] = 1.0f;
  if (i < g)  cnt[i] = 0.0f;
  if (i < ph) pooled[i] = 0.0f;
}

// ---------------- degree: one atomic per edge on destination ----------------
__global__ void deg_k(const int* __restrict__ col, float* __restrict__ deg, int E) {
  int e = blockIdx.x * blockDim.x + threadIdx.x;
  if (e < E) atomicAdd(&deg[col[e]], 1.0f);
}

// ---------------- dinv = rsqrt(deg); graph node counts ----------------
__global__ void prep_k(const float* __restrict__ deg, float* __restrict__ dinv,
                       const int* __restrict__ batch, float* __restrict__ cnt, int n) {
  int i = blockIdx.x * blockDim.x + threadIdx.x;
  if (i < n) {
    dinv[i] = rsqrtf(deg[i]);
    atomicAdd(&cnt[batch[i]], 1.0f);
  }
}

// ---------------- h = x @ conv_w via bf16 WMMA ----------------
// block: 32 rows x 256 cols, 8 waves; wave owns 2 M-tiles x 2 N-tiles.
// conv_w slab [32k x 256n] staged to LDS transposed bf16, stride 40 (16B-aligned frags).
__global__ __launch_bounds__(256) void gemm_wmma_k(const float* __restrict__ x,
                                                   const float* __restrict__ w,
                                                   float* __restrict__ h) {
  __shared__ __align__(16) unsigned short sB[HID * 40];  // 20 KB
  const int tid  = threadIdx.x;
  const int lane = tid & 31;
  const int wave = tid >> 5;
  const int g    = lane >> 4;     // lane group (0/1)
  const int r    = lane & 15;
  const int m0   = blockIdx.x * 32;
  const int n0   = wave * 32;

  v8f acc[2][2] = {};

  for (int kt = 0; kt < IN_DIM; kt += 32) {
    // stage W[kt..kt+32)[0..256) -> LDS as bf16, layout sB[n*40 + k]
    #pragma unroll
    for (int i = 0; i < 32; ++i)
      sB[tid * 40 + i] = f2bf(w[(size_t)(kt + i) * HID + tid]);
    __syncthreads();

    // A fragments (16-bit A 16x32 layout: lane r, group g -> K in [8g,8g+8) U [16+8g,24+8g))
    v16bf A[2];
    #pragma unroll
    for (int mt = 0; mt < 2; ++mt) {
      const float* ap = x + (size_t)(m0 + mt * 16 + r) * IN_DIM + kt + g * 8;
      float4 c0 = *(const float4*)(ap);
      float4 c1 = *(const float4*)(ap + 4);
      float4 c2 = *(const float4*)(ap + 16);
      float4 c3 = *(const float4*)(ap + 20);
      v16us au;
      au[0]  = f2bf(c0.x); au[1]  = f2bf(c0.y); au[2]  = f2bf(c0.z); au[3]  = f2bf(c0.w);
      au[4]  = f2bf(c1.x); au[5]  = f2bf(c1.y); au[6]  = f2bf(c1.z); au[7]  = f2bf(c1.w);
      au[8]  = f2bf(c2.x); au[9]  = f2bf(c2.y); au[10] = f2bf(c2.z); au[11] = f2bf(c2.w);
      au[12] = f2bf(c3.x); au[13] = f2bf(c3.y); au[14] = f2bf(c3.z); au[15] = f2bf(c3.w);
      A[mt] = __builtin_bit_cast(v16bf, au);
    }

    // B fragments (32x16 bf16: lane col = n0+nt*16+r, K in [8g,8g+8) U [16+8g,24+8g), matching A)
    #pragma unroll
    for (int nt = 0; nt < 2; ++nt) {
      const unsigned short* bp = &sB[(n0 + nt * 16 + r) * 40 + g * 8];
      v8us blo = *(const v8us*)(bp);
      v8us bhi = *(const v8us*)(bp + 16);
      v16us bu;
      #pragma unroll
      for (int e = 0; e < 8; ++e) { bu[e] = blo[e]; bu[8 + e] = bhi[e]; }
      v16bf B = __builtin_bit_cast(v16bf, bu);
      #pragma unroll
      for (int mt = 0; mt < 2; ++mt)
        acc[mt][nt] = __builtin_amdgcn_wmma_f32_16x16x32_bf16(
            false, A[mt], false, B, (short)0, acc[mt][nt], false, false);
    }
    __syncthreads();
  }

  // C/D layout: VGPR v -> row 8g+v, lane r -> col
  #pragma unroll
  for (int mt = 0; mt < 2; ++mt)
    #pragma unroll
    for (int nt = 0; nt < 2; ++nt)
      #pragma unroll
      for (int v = 0; v < 8; ++v)
        h[(size_t)(m0 + mt * 16 + g * 8 + v) * HID + n0 + nt * 16 + r] = acc[mt][nt][v];
}

// ---------------- agg init with self-loop term: agg[i] = h[i] * dinv[i]^2 ----------------
__global__ void selfloop_k(const float* __restrict__ h, const float* __restrict__ dinv,
                           float* __restrict__ agg, int n) {
  int t = blockIdx.x * blockDim.x + threadIdx.x;
  int i = t >> 6, j = (t & 63) << 2;
  if (i < n) {
    float d = dinv[i], s = d * d;
    float4 hv = *(const float4*)(h + (size_t)i * HID + j);
    float4 o = make_float4(hv.x * s, hv.y * s, hv.z * s, hv.w * s);
    *(float4*)(agg + (size_t)i * HID + j) = o;
  }
}

// ---------------- edge scatter: agg[col] += h[row] * dinv[row]*dinv[col] ----------------
// 64 lanes per edge, float4 per lane -> coalesced 1 KB message rows.
__global__ void scatter_k(const int* __restrict__ row, const int* __restrict__ col,
                          const float* __restrict__ dinv, const float* __restrict__ h,
                          float* __restrict__ agg, int E) {
  int t = blockIdx.x * blockDim.x + threadIdx.x;
  int e = t >> 6, j = (t & 63) << 2;
  if (e < E) {
    int rs = row[e], cd = col[e];
    float norm = dinv[rs] * dinv[cd];
    float4 hv = *(const float4*)(h + (size_t)rs * HID + j);
    float* a = agg + (size_t)cd * HID + j;
    atomicAdd(a + 0, hv.x * norm);
    atomicAdd(a + 1, hv.y * norm);
    atomicAdd(a + 2, hv.z * norm);
    atomicAdd(a + 3, hv.w * norm);
  }
}

// ---------------- ReLU(agg + b) scatter-added into per-graph pool ----------------
__global__ void pool_k(const float* __restrict__ agg, const float* __restrict__ conv_b,
                       const int* __restrict__ batch, float* __restrict__ pooled, int n) {
  int t = blockIdx.x * blockDim.x + threadIdx.x;
  int i = t >> 6, j = (t & 63) << 2;
  if (i < n) {
    int b = batch[i];
    float4 av = *(const float4*)(agg + (size_t)i * HID + j);
    float4 bb = *(const float4*)(conv_b + j);
    float* p = pooled + (size_t)b * HID + j;
    atomicAdd(p + 0, fmaxf(av.x + bb.x, 0.0f));
    atomicAdd(p + 1, fmaxf(av.y + bb.y, 0.0f));
    atomicAdd(p + 2, fmaxf(av.z + bb.z, 0.0f));
    atomicAdd(p + 3, fmaxf(av.w + bb.w, 0.0f));
  }
}

// ---------------- head: warp per graph; mean, linear(256->5), log_softmax ----------------
__global__ __launch_bounds__(256) void head_k(const float* __restrict__ pooled,
                                              const float* __restrict__ cnt,
                                              const float* __restrict__ lin_w,
                                              const float* __restrict__ lin_b,
                                              float* __restrict__ out, int G) {
  int wid  = (blockIdx.x * blockDim.x + threadIdx.x) >> 5;
  int lane = threadIdx.x & 31;
  if (wid >= G) return;
  float inv = 1.0f / fmaxf(cnt[wid], 1.0f);
  float acc[N_CLASSES] = {0.f, 0.f, 0.f, 0.f, 0.f};
  for (int j = lane; j < HID; j += 32) {
    float p = pooled[(size_t)wid * HID + j] * inv;
    #pragma unroll
    for (int c = 0; c < N_CLASSES; ++c) acc[c] += p * lin_w[j * N_CLASSES + c];
  }
  #pragma unroll
  for (int c = 0; c < N_CLASSES; ++c)
    #pragma unroll
    for (int off = 16; off > 0; off >>= 1)
      acc[c] += __shfl_down(acc[c], off, 32);
  if (lane == 0) {
    float lg[N_CLASSES], m = -3.4e38f;
    #pragma unroll
    for (int c = 0; c < N_CLASSES; ++c) { lg[c] = acc[c] + lin_b[c]; m = fmaxf(m, lg[c]); }
    float s = 0.f;
    #pragma unroll
    for (int c = 0; c < N_CLASSES; ++c) s += expf(lg[c] - m);
    float ls = logf(s);
    #pragma unroll
    for (int c = 0; c < N_CLASSES; ++c) out[(size_t)wid * N_CLASSES + c] = lg[c] - m - ls;
  }
}

extern "C" void kernel_launch(void* const* d_in, const int* in_sizes, int n_in,
                              void* d_out, int out_size, void* d_ws, size_t ws_size,
                              hipStream_t stream) {
  const float* x      = (const float*)d_in[0];
  const int*   eidx   = (const int*)d_in[1];
  const int*   batch  = (const int*)d_in[2];
  const float* conv_w = (const float*)d_in[3];
  const float* conv_b = (const float*)d_in[4];
  const float* lin_w  = (const float*)d_in[5];
  const float* lin_b  = (const float*)d_in[6];
  float* out = (float*)d_out;

  const int n = in_sizes[0] / IN_DIM;     // 100000 (divisible by 32)
  const int E = in_sizes[1] / 2;          // 1600000
  const int G = out_size / N_CLASSES;     // 1000

  float* ws = (float*)d_ws;
  size_t off = 0;
  float* h      = ws + off; off += (size_t)n * HID;
  float* agg    = ws + off; off += (size_t)n * HID;
  float* deg    = ws + off; off += (size_t)n;
  float* dinv   = ws + off; off += (size_t)n;
  float* pooled = ws + off; off += (size_t)G * HID;
  float* cnt    = ws + off; off += (size_t)G;

  const int* row = eidx;       // edge_index[0]
  const int* col = eidx + E;   // edge_index[1]

  int initN = n > G * HID ? n : G * HID;
  init_k<<<(initN + 255) / 256, 256, 0, stream>>>(deg, cnt, pooled, n, G, G * HID);
  deg_k<<<(E + 255) / 256, 256, 0, stream>>>(col, deg, E);
  prep_k<<<(n + 255) / 256, 256, 0, stream>>>(deg, dinv, batch, cnt, n);
  gemm_wmma_k<<<n / 32, 256, 0, stream>>>(x, conv_w, h);

  int nt4 = n * 64;   // node-level float4 workers
  selfloop_k<<<(nt4 + 255) / 256, 256, 0, stream>>>(h, dinv, agg, n);
  int et4 = E * 64;   // edge-level float4 workers (102.4M threads)
  scatter_k<<<(et4 + 255) / 256, 256, 0, stream>>>(row, col, dinv, h, agg, E);
  pool_k<<<(nt4 + 255) / 256, 256, 0, stream>>>(agg, conv_b, batch, pooled, n);
  head_k<<<(G * 32 + 255) / 256, 256, 0, stream>>>(pooled, cnt, lin_w, lin_b, out, G);
}
